// DirGCNConv_2_45535243272405
// MI455X (gfx1250) — matrix operand order repaired and checked by reading, not compiled, not run on hardware
//
#include <hip/hip_runtime.h>

#define N_NODES 50000
#define DFEAT   128
#define ALPHA   0.5f
#define BETA    0.5f
#define GAMA    0.5f
#define CHUNK   8          // edges per wave in SpMM

typedef __attribute__((ext_vector_type(16))) __bf16 v16bf;
typedef __attribute__((ext_vector_type(8)))  __bf16 v8bf;
typedef __attribute__((ext_vector_type(8)))  float  v8f;

__device__ __forceinline__ unsigned short f2bf_bits(float f) {
    // round-to-nearest-even f32 -> bf16 bit pattern
    unsigned u = __builtin_bit_cast(unsigned, f);
    unsigned r = (u + 0x7FFFu + ((u >> 16) & 1u)) >> 16;
    return (unsigned short)r;
}

// ---------------------------------------------------------------------------
// out[i][j] = sum of scaled biases
// ---------------------------------------------------------------------------
__global__ void init_out_kernel(float* __restrict__ out,
                                const float* __restrict__ bsd, const float* __restrict__ bds,
                                const float* __restrict__ b0,  const float* __restrict__ b1,
                                const float* __restrict__ b2,  const float* __restrict__ b3) {
    long idx = (long)blockIdx.x * blockDim.x + threadIdx.x;
    if (idx >= (long)N_NODES * DFEAT) return;
    int j = (int)(idx & (DFEAT - 1));
    out[idx] = ALPHA * bsd[j] + (1.0f - ALPHA) * bds[j]
             + BETA  * b0[j]  + (1.0f - BETA)  * b1[j]
             + GAMA  * b2[j]  + (1.0f - GAMA)  * b3[j];
}

// ---------------------------------------------------------------------------
// Zero the SpMM accumulator (float4 stores, grid-stride).
// ---------------------------------------------------------------------------
__global__ void zero_kernel(float4* __restrict__ p, long n4) {
    long i = (long)blockIdx.x * blockDim.x + threadIdx.x;
    long stride = (long)gridDim.x * blockDim.x;
    float4 z = make_float4(0.f, 0.f, 0.f, 0.f);
    for (; i < n4; i += stride) p[i] = z;
}

// ---------------------------------------------------------------------------
// f32 -> bf16 pack (4 elements / thread, grid-stride).
// ---------------------------------------------------------------------------
__global__ void cvt_bf16_kernel(const float4* __restrict__ in, ushort4* __restrict__ outp, long n4) {
    long i = (long)blockIdx.x * blockDim.x + threadIdx.x;
    long stride = (long)gridDim.x * blockDim.x;
    for (; i < n4; i += stride) {
        float4 f = in[i];
        ushort4 o;
        o.x = f2bf_bits(f.x); o.y = f2bf_bits(f.y);
        o.z = f2bf_bits(f.z); o.w = f2bf_bits(f.w);
        outp[i] = o;
    }
}

// ---------------------------------------------------------------------------
// SpMM: y[rows[e]] += vals[e] * x[cols[e]]
// One wave32 per CHUNK consecutive edges; COO is row-sorted (np.unique key =
// row*n+col), so register-accumulate while row is constant and flush with
// native f32 atomics only on row change / chunk end. Lane l owns features
// 4l..4l+3 (32 lanes x float4 = 128 features).
// ---------------------------------------------------------------------------
__global__ void spmm_kernel(const int* __restrict__ rows, const int* __restrict__ cols,
                            const float* __restrict__ vals, const float* __restrict__ x,
                            float* __restrict__ y, int nnz) {
    long wid  = ((long)blockIdx.x * blockDim.x + threadIdx.x) >> 5;
    int  lane = threadIdx.x & 31;
    long e0 = wid * CHUNK;
    if (e0 >= nnz) return;
    long e1 = e0 + CHUNK; if (e1 > nnz) e1 = nnz;

    float4 acc = make_float4(0.f, 0.f, 0.f, 0.f);
    int cur_r = rows[e0];
    for (long e = e0; e < e1; ++e) {
        int   r = rows[e];
        int   c = cols[e];
        float v = vals[e];
        if (r != cur_r) {                       // wave-uniform branch
            float* yp = y + (size_t)cur_r * DFEAT + lane * 4;
            unsafeAtomicAdd(yp + 0, acc.x);
            unsafeAtomicAdd(yp + 1, acc.y);
            unsafeAtomicAdd(yp + 2, acc.z);
            unsafeAtomicAdd(yp + 3, acc.w);
            acc = make_float4(0.f, 0.f, 0.f, 0.f);
            cur_r = r;
        }
        float4 xv = ((const float4*)(x + (size_t)c * DFEAT))[lane];
        acc.x = fmaf(v, xv.x, acc.x);
        acc.y = fmaf(v, xv.y, acc.y);
        acc.z = fmaf(v, xv.z, acc.z);
        acc.w = fmaf(v, xv.w, acc.w);
    }
    float* yp = y + (size_t)cur_r * DFEAT + lane * 4;
    unsafeAtomicAdd(yp + 0, acc.x);
    unsafeAtomicAdd(yp + 1, acc.y);
    unsafeAtomicAdd(yp + 2, acc.z);
    unsafeAtomicAdd(yp + 3, acc.w);
}

// ---------------------------------------------------------------------------
// GEMM epilogue with WMMA: out += scale * (Ybf @ Wbf), bf16 inputs, f32 acc.
// Block = 256 threads = 8 waves; block b owns row-tile b (16 rows; 50000 =
// 3125*16 exactly); wave w owns 16-column tile w. K = 128 = 4 x 32.
// Fragment layouts (ISA 7.12.2, wave32):
//   A 16x32: lane -> M = lane&15; K-chunks [half*8,+8) and [16+half*8,+8)
//   B 32x16: lane -> N = lane&15; same K-chunks as A
//   C/D:     lane L, vgpr r -> (M = r + (L>=16)*8, N = L&15)
// ---------------------------------------------------------------------------
__global__ void gemm_acc_kernel(const __bf16* __restrict__ Ybf, const __bf16* __restrict__ Wbf,
                                float* __restrict__ out, float scale) {
    const int lane   = threadIdx.x & 31;
    const int tn     = threadIdx.x >> 5;
    const int tile_m = blockIdx.x;
    const int half   = lane >> 4;
    const int mloc   = lane & 15;
    const int n0     = tn << 4;

    const __bf16* yrow = Ybf + (size_t)(tile_m * 16 + mloc) * DFEAT;

    v8f acc = {};
#pragma unroll
    for (int kt = 0; kt < 4; ++kt) {
        const int kbase = kt * 32;

        v8bf alo = *(const v8bf*)(yrow + kbase + half * 8);        // 16B load
        v8bf ahi = *(const v8bf*)(yrow + kbase + 16 + half * 8);   // 16B load
        v16bf a = __builtin_shufflevector(alo, ahi,
                    0, 1, 2, 3, 4, 5, 6, 7, 8, 9, 10, 11, 12, 13, 14, 15);

        const __bf16* wcol = Wbf + (size_t)kbase * DFEAT + n0 + mloc;
        v16bf b;
#pragma unroll
        for (int s = 0; s < 16; ++s) {
            const int kk = (s < 8) ? (half * 8 + s) : (16 + half * 8 + (s - 8));
            b[s] = wcol[(size_t)kk * DFEAT];
        }

        acc = __builtin_amdgcn_wmma_f32_16x16x32_bf16(
            false, a, false, b, (short)0, acc, false, false);
    }

#pragma unroll
    for (int r = 0; r < 8; ++r) {
        int om = tile_m * 16 + half * 8 + r;
        float* op = out + (size_t)om * DFEAT + n0 + mloc;
        *op += scale * acc[r];
    }
}

// ---------------------------------------------------------------------------
// Inputs: 0:x  1..12: (w,b) x6   13..30: {rows,cols,vals} x {a,at,aat,ata,aa,atat}
// ---------------------------------------------------------------------------
extern "C" void kernel_launch(void* const* d_in, const int* in_sizes, int n_in,
                              void* d_out, int out_size, void* d_ws, size_t ws_size,
                              hipStream_t stream) {
    const float* x = (const float*)d_in[0];

    const float* w[6] = {(const float*)d_in[1], (const float*)d_in[3],
                         (const float*)d_in[5], (const float*)d_in[7],
                         (const float*)d_in[9], (const float*)d_in[11]};
    const float* b[6] = {(const float*)d_in[2], (const float*)d_in[4],
                         (const float*)d_in[6], (const float*)d_in[8],
                         (const float*)d_in[10], (const float*)d_in[12]};
    const int base[6] = {13, 16, 19, 22, 25, 28};
    const float scale[6] = {ALPHA, 1.0f - ALPHA, BETA, 1.0f - BETA, GAMA, 1.0f - GAMA};

    float* out = (float*)d_out;

    const long total = (long)N_NODES * DFEAT;                 // 6.4M
    float*   Y   = (float*)d_ws;                              // 25.6 MB
    __bf16*  Ybf = (__bf16*)((char*)d_ws + total * 4);        // 12.8 MB
    __bf16*  Wbf = (__bf16*)((char*)d_ws + total * 6);        // 32 KB

    init_out_kernel<<<(int)((total + 255) / 256), 256, 0, stream>>>(
        out, b[0], b[1], b[2], b[3], b[4], b[5]);

    for (int k = 0; k < 6; ++k) {
        const int*   rows = (const int*)d_in[base[k] + 0];
        const int*   cols = (const int*)d_in[base[k] + 1];
        const float* vals = (const float*)d_in[base[k] + 2];
        const int    nnz  = in_sizes[base[k] + 2];

        zero_kernel<<<2048, 256, 0, stream>>>((float4*)Y, total / 4);

        // 8 waves per block, CHUNK edges per wave
        long waves = ((long)nnz + CHUNK - 1) / CHUNK;
        int  spmm_blocks = (int)((waves + 7) / 8);
        spmm_kernel<<<spmm_blocks, 256, 0, stream>>>(rows, cols, vals, x, Y, nnz);

        cvt_bf16_kernel<<<2048, 256, 0, stream>>>((const float4*)Y, (ushort4*)Ybf, total / 4);
        cvt_bf16_kernel<<<16, 256, 0, stream>>>((const float4*)w[k], (ushort4*)Wbf,
                                                (long)(DFEAT * DFEAT / 4));

        gemm_acc_kernel<<<N_NODES / 16, 256, 0, stream>>>(Ybf, Wbf, out, scale[k]);
    }
}
